// DeformableCore_51247549776082
// MI455X (gfx1250) — matrix operand order repaired and checked by reading, not compiled, not run on hardware
//
#include <hip/hip_runtime.h>

#define BB   2
#define CC   256
#define NH_  8
#define NP_  4
#define NL_  5
#define DH_  32
#define TOK  5456
#define MTOT (BB*TOK)   /* 10912 */
#define FFN_ 1024
#define NLAY 6

typedef __bf16 bf16;
typedef __attribute__((ext_vector_type(16))) __bf16 v16bf;
typedef __attribute__((ext_vector_type(8)))  __bf16 v8bf;
typedef __attribute__((ext_vector_type(8)))  float  v8f;
typedef unsigned int v4u __attribute__((ext_vector_type(4)));
typedef int          v8i __attribute__((ext_vector_type(8)));
typedef int          v4i __attribute__((ext_vector_type(4)));

// ---- LDS byte-offset of a __shared__ pointer (generic -> AS(3) -> int) ----
__device__ __forceinline__ unsigned lds_addr_of(const void* p)
{
    return (unsigned)(unsigned long long)(__attribute__((address_space(3))) const void*)p;
}

// ---------------------------------------------------------------------------
// TDM 2D tile load: 64 rows x 32-element (64B) bf16 tile from a row-major
// matrix with row stride ldk (elements).  tensor_dim0/1 are set to the
// REMAINING extent from the tile origin so out-of-matrix reads return 0
// (handles ragged M/N edges with no masking).  D# per CDNA5 ISA ch.8:
//   group0: count=1 | lds_addr | global_addr(57b) | type=2
//   group1: data_size=1(2B), tensor_dim0/1, tile_dim0=32, tile_dim1=64,
//           tensor_dim0_stride=ldk
// Builtin signature (clang-23 / amdgpu-toolchain): 6 args
//   (v4u g0, v8i g1, v4i g2, v4i g3, v8i g4, i32 cpol)
// ---------------------------------------------------------------------------
__device__ __forceinline__ void tdm_load_2d(unsigned lds, const bf16* gptr,
                                            int rows_rem, int k_rem, int ldk)
{
    unsigned long long ga = (unsigned long long)gptr;
    v4u g0;
    g0.x = 1u;                                  // count=1 (valid descriptor)
    g0.y = lds;                                 // LDS byte address
    g0.z = (unsigned)ga;                        // global_addr[31:0]
    g0.w = ((unsigned)(ga >> 32) & 0x01FFFFFFu) // global_addr[56:32]
         | 0x80000000u;                         // type=2 ("image") in [127:126]
    unsigned td0 = (unsigned)k_rem;             // tensor_dim0 (elements)
    unsigned td1 = (unsigned)rows_rem;          // tensor_dim1 (rows)
    v8i g1;
    g1[0] = (int)0x00010000u;                   // data_size=1 (2 bytes), mask=0
    g1[1] = (int)((td0 & 0xFFFFu) << 16);       // tensor_dim0[15:0] @ bits 63:48
    g1[2] = (int)(((td0 >> 16) & 0xFFFFu) | ((td1 & 0xFFFFu) << 16));
    g1[3] = (int)(((td1 >> 16) & 0xFFFFu) | (32u << 16));   // tile_dim0 = 32
    g1[4] = 64;                                 // tile_dim1 = 64, tile_dim2 = 0
    g1[5] = ldk;                                // tensor_dim0_stride[31:0]
    g1[6] = 0;                                  // stride hi / dim1_stride lo
    g1[7] = 0;
    v4i z4 = {0, 0, 0, 0};
    v8i z8 = {0, 0, 0, 0, 0, 0, 0, 0};
    __builtin_amdgcn_tensor_load_to_lds(g0, g1, z4, z4, z8, 0);
}

// ---------------------------------------------------------------------------
// WMMA GEMM, TDM/LDS-staged:  Out[m,n] = sum_k A[m,k]*Bt[n,k] + bias[n]
// A:[M,K] bf16 row-major; Bt:[N,K] bf16 (weights pre-transposed).
// Workgroup = 4 waves (2x2), 64x64 output tile; each wave 32x32 via 2x2
// v_wmma_f32_16x16x32_bf16.  Double-buffered LDS tiles filled by TDM
// (wave 0 issues, s_wait_tensorcnt + barriers synchronize).
// Epilogue: bias, optional ReLU / residual-add, fp32 or bf16 store, with
// (batch,level)->token row scatter: orow = obase + (m/rowsPerB)*obstride + m%rowsPerB
// ---------------------------------------------------------------------------
#define WMMA_BF16(a, b, c) \
    __builtin_amdgcn_wmma_f32_16x16x32_bf16(false, (a), false, (b), (short)0, (c), false, false)

template<bool WRITE_BF16, bool RELU, bool RESID>
__global__ void __launch_bounds__(128)
k_gemm64(const bf16* __restrict__ A, const bf16* __restrict__ Bt,
         const float* __restrict__ bias,
         float* __restrict__ Of, bf16* __restrict__ Ob,
         const float* __restrict__ Rf,
         int M, int Nn, int K, int rowsPerB, int obase, int obstride, int ldo)
{
    __shared__ __align__(128) bf16 sA[2][64 * 32];
    __shared__ __align__(128) bf16 sB[2][64 * 32];

    const int tn = blockIdx.x, tm = blockIdx.y;
    const int w    = threadIdx.y;          // wave id 0..3
    const int wy   = w >> 1, wx = w & 1;
    const int lane = threadIdx.x;
    const int half = lane >> 4;
    const int l15  = lane & 15;

    const int m0g = tm * 64;
    const int n0g = tn * 64;
    const int nkt = K / 32;

    if (w == 0) {   // prologue: fill buffer 0
        tdm_load_2d(lds_addr_of(&sA[0][0]), A  + (size_t)m0g * K, M  - m0g, K, K);
        tdm_load_2d(lds_addr_of(&sB[0][0]), Bt + (size_t)n0g * K, Nn - n0g, K, K);
    }

    v8f acc00 = {}, acc01 = {}, acc10 = {}, acc11 = {};

    for (int kt = 0; kt < nkt; kt++) {
        if (w == 0) {
            if (kt + 1 < nkt) {
                int k0n = (kt + 1) * 32;
                tdm_load_2d(lds_addr_of(&sA[(kt + 1) & 1][0]),
                            A  + (size_t)m0g * K + k0n, M  - m0g, K - k0n, K);
                tdm_load_2d(lds_addr_of(&sB[(kt + 1) & 1][0]),
                            Bt + (size_t)n0g * K + k0n, Nn - n0g, K - k0n, K);
                __builtin_amdgcn_s_wait_tensorcnt(2);  // current buffer done
            } else {
                __builtin_amdgcn_s_wait_tensorcnt(0);
            }
        }
        __syncthreads();                               // publish LDS tiles

        const bf16* pa = &sA[kt & 1][0];
        const bf16* pb = &sB[kt & 1][0];

        // A fragments (rows wy*32 + {0,16}), B fragments (cols wx*32 + {0,16})
        v16bf af[2], bfv[2];
        #pragma unroll
        for (int t = 0; t < 2; t++) {
            const bf16* p = pa + (wy * 32 + t * 16 + l15) * 32 + half * 8;
            v8bf a0 = *(const v8bf*)(p);
            v8bf a1 = *(const v8bf*)(p + 16);
            af[t] = __builtin_shufflevector(a0, a1,
                        0,1,2,3,4,5,6,7, 8,9,10,11,12,13,14,15);
            const bf16* q = pb + (wx * 32 + t * 16 + l15) * 32 + half * 8;
            v8bf b0 = *(const v8bf*)(q);
            v8bf b1 = *(const v8bf*)(q + 16);
            bfv[t] = __builtin_shufflevector(b0, b1,
                        0,1,2,3,4,5,6,7, 8,9,10,11,12,13,14,15);
        }
        acc00 = WMMA_BF16(af[0], bfv[0], acc00);
        acc01 = WMMA_BF16(af[0], bfv[1], acc01);
        acc10 = WMMA_BF16(af[1], bfv[0], acc10);
        acc11 = WMMA_BF16(af[1], bfv[1], acc11);

        __syncthreads();                               // done reading this buffer
    }

    // ---- epilogue -------------------------------------------------------
    auto epi = [&](v8f acc, int mt, int nt) {
        int col = n0g + wx * 32 + nt * 16 + l15;
        if (col >= Nn) return;
        float bv_ = bias[col];
        #pragma unroll
        for (int r = 0; r < 8; r++) {
            int mrow = m0g + wy * 32 + mt * 16 + half * 8 + r;
            if (mrow >= M) continue;
            float v = acc[r] + bv_;
            if (RELU) v = v > 0.f ? v : 0.f;
            size_t orow = (size_t)obase + (size_t)(mrow / rowsPerB) * obstride
                        + (size_t)(mrow % rowsPerB);
            size_t oidx = orow * (size_t)ldo + col;
            if (RESID) v += Rf[oidx];
            if (WRITE_BF16) Ob[oidx] = (bf16)v;
            else            Of[oidx] = v;
        }
    };
    epi(acc00, 0, 0); epi(acc01, 0, 1); epi(acc10, 1, 0); epi(acc11, 1, 1);
}

// ---- input transpose: x[b,k,hw] -> xt[(b*HW+hw), k] bf16 ------------------
__global__ void k_transpose_in(const float* __restrict__ x, bf16* __restrict__ xt,
                               int Cin, int HW)
{
    long long tot = (long long)BB * HW * Cin;
    long long idx = (long long)blockIdx.x * blockDim.x + threadIdx.x;
    if (idx >= tot) return;
    int k  = (int)(idx % Cin);
    long long m = idx / Cin;
    int b  = (int)(m / HW);
    int hw = (int)(m % HW);
    xt[idx] = (bf16)x[((size_t)(b * Cin + k)) * HW + hw];
}

// ---- weight convert (already [N][K] layout, e.g. 1x1 conv weights) --------
__global__ void k_convD(const float* __restrict__ w, bf16* __restrict__ wt, int n)
{
    int i = blockIdx.x * blockDim.x + threadIdx.x;
    if (i < n) wt[i] = (bf16)w[i];
}

// ---- weight convert+transpose: w[K,N] -> wt[N][K] bf16 --------------------
__global__ void k_convT(const float* __restrict__ w, bf16* __restrict__ wt,
                        int K, int Nn)
{
    int i = blockIdx.x * blockDim.x + threadIdx.x;
    if (i >= K * Nn) return;
    int k  = i % K;
    int nn = i / K;
    wt[i] = (bf16)w[(size_t)k * Nn + nn];
}

// ---- direct 3x3 stride-2 pad-1 conv (small: 8x8 and 4x4 outputs) ----------
__global__ void k_conv3x3s2(const float* __restrict__ x, const float* __restrict__ w,
                            const float* __restrict__ bias, float* __restrict__ y,
                            int Cin, int Hin, int Hout)
{
    int idx = blockIdx.x * blockDim.x + threadIdx.x;
    int total = BB * CC * Hout * Hout;
    if (idx >= total) return;
    int ox = idx % Hout;
    int oy = (idx / Hout) % Hout;
    int co = (idx / (Hout * Hout)) % CC;
    int b  = idx / (Hout * Hout * CC);
    float acc = bias[co];
    for (int ci = 0; ci < Cin; ci++) {
        const float* xp = x + ((size_t)(b * Cin + ci)) * Hin * Hin;
        const float* wp = w + ((size_t)(co * Cin + ci)) * 9;
        #pragma unroll
        for (int ky = 0; ky < 3; ky++) {
            int ys = oy * 2 - 1 + ky;
            if (ys < 0 || ys >= Hin) continue;
            #pragma unroll
            for (int kx = 0; kx < 3; kx++) {
                int xs = ox * 2 - 1 + kx;
                if (xs < 0 || xs >= Hin) continue;
                acc += xp[ys * Hin + xs] * wp[ky * 3 + kx];
            }
        }
    }
    y[idx] = acc;
}

// ---- GroupNorm(8) + ReLU on [B,C,HW]; one block per (b,group) -------------
__global__ void k_gn_relu(const float* __restrict__ x, const float* __restrict__ g,
                          const float* __restrict__ bta, float* __restrict__ y, int HW)
{
    int b  = blockIdx.x / 8;
    int gr = blockIdx.x % 8;
    __shared__ float ss[64], ss2[64];
    int t = threadIdx.x;
    int cnt = 32 * HW;
    float s = 0.f, s2 = 0.f;
    for (int i = t; i < cnt; i += 64) {
        int ci = gr * 32 + i / HW;
        int hw = i % HW;
        float v = x[((size_t)(b * CC + ci)) * HW + hw];
        s += v; s2 += v * v;
    }
    ss[t] = s; ss2[t] = s2; __syncthreads();
    for (int off = 32; off > 0; off >>= 1) {
        if (t < off) { ss[t] += ss[t + off]; ss2[t] += ss2[t + off]; }
        __syncthreads();
    }
    float mean = ss[0] / cnt;
    float var  = ss2[0] / cnt - mean * mean;
    float inv  = rsqrtf(var + 1e-5f);
    for (int i = t; i < cnt; i += 64) {
        int ci = gr * 32 + i / HW;
        int hw = i % HW;
        size_t id = ((size_t)(b * CC + ci)) * HW + hw;
        float v = (x[id] - mean) * inv * g[ci] + bta[ci];
        y[id] = v > 0.f ? v : 0.f;
    }
}

// ---- scatter conv map [B,C,HW] into token tensor [B*TOK, C] ---------------
__global__ void k_scatter(const float* __restrict__ m, float* __restrict__ feats,
                          int HW, int start)
{
    int idx = blockIdx.x * blockDim.x + threadIdx.x;
    int total = BB * CC * HW;
    if (idx >= total) return;
    int hw = idx % HW;
    int c  = (idx / HW) % CC;
    int b  = idx / (HW * CC);
    feats[((size_t)(b * TOK + start + hw)) * CC + c] = m[idx];
}

// ---- LayerNorm over C=256; one wave per token row; writes bf16 ------------
__global__ void __launch_bounds__(32)
k_layernorm(const float* __restrict__ X, const float* __restrict__ g,
            const float* __restrict__ b, bf16* __restrict__ Y)
{
    int r = blockIdx.x;
    int lane = threadIdx.x;
    const float* x = X + (size_t)r * CC;
    float v[8];
    float s = 0.f, s2 = 0.f;
    #pragma unroll
    for (int j = 0; j < 8; j++) {
        v[j] = x[lane * 8 + j];
        s += v[j]; s2 += v[j] * v[j];
    }
    #pragma unroll
    for (int off = 16; off > 0; off >>= 1) {
        s  += __shfl_xor(s,  off, 32);
        s2 += __shfl_xor(s2, off, 32);
    }
    float mean = s / 256.f;
    float var  = s2 / 256.f - mean * mean;
    float inv  = rsqrtf(var + 1e-5f);
    #pragma unroll
    for (int j = 0; j < 8; j++) {
        int c = lane * 8 + j;
        Y[(size_t)r * CC + c] = (bf16)(((v[j] - mean) * inv) * g[c] + b[c]);
    }
}

// ---- softmax over NL*NP=20 per (token,head), in place ---------------------
__global__ void k_softmax20(float* __restrict__ att)
{
    int idx = blockIdx.x * blockDim.x + threadIdx.x;
    if (idx >= MTOT * NH_) return;
    int r = idx / NH_, h = idx % NH_;
    float* p = att + (size_t)r * (NH_ * NL_ * NP_) + h * (NL_ * NP_);
    float mx = p[0];
    #pragma unroll
    for (int j = 1; j < 20; j++) mx = fmaxf(mx, p[j]);
    float e[20]; float s = 0.f;
    #pragma unroll
    for (int j = 0; j < 20; j++) { e[j] = expf(p[j] - mx); s += e[j]; }
    float inv = 1.f / s;
    #pragma unroll
    for (int j = 0; j < 20; j++) p[j] = e[j] * inv;
}

// ---- deformable sampling: one wave per (b,n,h), lane = d in head ----------
__global__ void __launch_bounds__(32)
k_sample(const float* __restrict__ off, const float* __restrict__ att,
         const float* __restrict__ val, bf16* __restrict__ out)
{
    int lane = threadIdx.x;
    int gid  = blockIdx.x;
    int h = gid % NH_;
    int r = gid / NH_;            // b*TOK + n
    int b = r / TOK;
    int n = r % TOK;

    const int dims[5]   = {64, 32, 16, 8, 4};
    const int sizes[5]  = {4096, 1024, 256, 64, 16};
    const int starts[5] = {0, 4096, 5120, 5376, 5440};

    int rem = n, lt = 0;
    while (lt < 4 && rem >= sizes[lt]) { rem -= sizes[lt]; lt++; }
    float ft = (float)dims[lt];
    int iy = rem / dims[lt], ix = rem % dims[lt];
    float px = (ix + 0.5f) / ft;
    float py = (iy + 0.5f) / ft;

    const float* offr = off + (size_t)r * (NH_ * NL_ * NP_ * 2);
    const float* attr = att + (size_t)r * (NH_ * NL_ * NP_) + h * (NL_ * NP_);

    float acc = 0.f;
    for (int l = 0; l < NL_; l++) {
        int fs = dims[l];
        float fsf = (float)fs;
        const float* vl = val + (size_t)(b * TOK + starts[l]) * CC + h * DH_ + lane;
        #pragma unroll
        for (int p = 0; p < NP_; p++) {
            int oc = ((h * NL_ + l) * NP_ + p) * 2;
            float lx = px + offr[oc]     / fsf;
            float ly = py + offr[oc + 1] / fsf;
            float x  = lx * fsf - 0.5f;
            float y  = ly * fsf - 0.5f;
            float x0 = floorf(x), y0 = floorf(y);
            float fx = x - x0, fy = y - y0;
            float a  = attr[l * NP_ + p];
            float s  = 0.f;
            #pragma unroll
            for (int cy = 0; cy < 2; cy++) {
                #pragma unroll
                for (int cx = 0; cx < 2; cx++) {
                    float xi = x0 + cx, yi = y0 + cy;
                    float wgt = (cx ? fx : 1.f - fx) * (cy ? fy : 1.f - fy);
                    bool inb = (xi >= 0.f) && (xi <= fsf - 1.f) &&
                               (yi >= 0.f) && (yi <= fsf - 1.f);
                    int xc = (int)fminf(fmaxf(xi, 0.f), fsf - 1.f);
                    int yc = (int)fminf(fmaxf(yi, 0.f), fsf - 1.f);
                    float g = vl[(size_t)(yc * fs + xc) * CC];
                    s += inb ? wgt * g : 0.f;
                }
            }
            acc += a * s;
        }
    }
    out[(size_t)r * CC + h * DH_ + lane] = (bf16)acc;
}

// ---- final un-concat: feats[B*TOK,C] -> 5 maps [B,C,fh,fw] flat -----------
__global__ void k_out(const float* __restrict__ feats, float* __restrict__ out)
{
    const int hws[5]    = {4096, 1024, 256, 64, 16};
    const int starts[5] = {0, 4096, 5120, 5376, 5440};
    int total = BB * CC * TOK;
    for (int idx = blockIdx.x * blockDim.x + threadIdx.x; idx < total;
         idx += gridDim.x * blockDim.x) {
        int rem = idx, l = 0;
        while (l < 4 && rem >= BB * CC * hws[l]) { rem -= BB * CC * hws[l]; l++; }
        int hw = rem % hws[l];
        int c  = (rem / hws[l]) % CC;
        int b  = rem / (hws[l] * CC);
        out[idx] = feats[((size_t)(b * TOK + starts[l] + hw)) * CC + c];
    }
}

// ===========================================================================
extern "C" void kernel_launch(void* const* d_in, const int* in_sizes, int n_in,
                              void* d_out, int out_size, void* d_ws, size_t ws_size,
                              hipStream_t stream)
{
    (void)in_sizes; (void)n_in; (void)out_size; (void)ws_size;
    const float* in0  = (const float*)d_in[0];
    const float* in1  = (const float*)d_in[1];
    const float* in2  = (const float*)d_in[2];
    const float* pw[3] = {(const float*)d_in[3], (const float*)d_in[5], (const float*)d_in[7]};
    const float* pb[3] = {(const float*)d_in[4], (const float*)d_in[6], (const float*)d_in[8]};
    const float* buw0 = (const float*)d_in[9];  const float* bub0 = (const float*)d_in[10];
    const float* gng  = (const float*)d_in[11]; const float* gnb  = (const float*)d_in[12];
    const float* buw1 = (const float*)d_in[13]; const float* bub1 = (const float*)d_in[14];
    const float* ln1g = (const float*)d_in[15]; const float* ln1b = (const float*)d_in[16];
    const float* woff = (const float*)d_in[17]; const float* boff = (const float*)d_in[18];
    const float* watt = (const float*)d_in[19]; const float* batt = (const float*)d_in[20];
    const float* wval = (const float*)d_in[21]; const float* bval = (const float*)d_in[22];
    const float* wout = (const float*)d_in[23]; const float* bout = (const float*)d_in[24];
    const float* ln2g = (const float*)d_in[25]; const float* ln2b = (const float*)d_in[26];
    const float* w1   = (const float*)d_in[27]; const float* b1   = (const float*)d_in[28];
    const float* w2   = (const float*)d_in[29]; const float* b2   = (const float*)d_in[30];

    char* wsp = (char*)d_ws;
    auto alloc = [&](size_t bytes) -> void* {
        void* p = (void*)wsp;
        wsp += (bytes + 255) & ~((size_t)255);
        return p;
    };
    float* feats = (float*)alloc((size_t)MTOT * CC * 4);
    bf16*  qbf   = (bf16*) alloc((size_t)MTOT * CC * 2);
    float* offb  = (float*)alloc((size_t)MTOT * 320 * 4);
    float* attb  = (float*)alloc((size_t)MTOT * 160 * 4);
    float* valb  = (float*)alloc((size_t)MTOT * CC * 4);
    bf16*  sampb = (bf16*) alloc((size_t)MTOT * CC * 2);
    bf16*  hidb  = (bf16*) alloc((size_t)MTOT * FFN_ * 2);
    bf16*  x0t   = (bf16*) alloc((size_t)BB * 4096 * 512 * 2);
    bf16*  x1t   = (bf16*) alloc((size_t)BB * 1024 * 1024 * 2);
    bf16*  x2t   = (bf16*) alloc((size_t)BB * 256 * 2048 * 2);
    bf16*  wbt   = (bf16*) alloc((size_t)2048 * 256 * 2);
    float* m0    = (float*)alloc((size_t)BB * CC * 64 * 4);
    float* g0    = (float*)alloc((size_t)BB * CC * 64 * 4);
    float* m1    = (float*)alloc((size_t)BB * CC * 16 * 4);

    const dim3 gb(32, 4);   // 4 waves per workgroup
    auto tiles = [](int x) { return (unsigned)((x + 63) / 64); };

    // ---- input pyramid: 1x1 conv projections as TDM+WMMA GEMMs ------------
    { long long t = (long long)BB * 4096 * 512;
      k_transpose_in<<<(unsigned)((t + 255) / 256), 256, 0, stream>>>(in0, x0t, 512, 4096); }
    { long long t = (long long)BB * 1024 * 1024;
      k_transpose_in<<<(unsigned)((t + 255) / 256), 256, 0, stream>>>(in1, x1t, 1024, 1024); }
    { long long t = (long long)BB * 256 * 2048;
      k_transpose_in<<<(unsigned)((t + 255) / 256), 256, 0, stream>>>(in2, x2t, 2048, 256); }

    const int  cins[3]   = {512, 1024, 2048};
    const int  hwsl[3]   = {4096, 1024, 256};
    const int  startl[3] = {0, 4096, 5120};
    bf16* xts[3] = {x0t, x1t, x2t};
    for (int l = 0; l < 3; l++) {
        int tot = CC * cins[l];
        k_convD<<<(tot + 255) / 256, 256, 0, stream>>>(pw[l], wbt, tot);
        int M = BB * hwsl[l];
        k_gemm64<false, false, false><<<dim3(tiles(CC), tiles(M)), gb, 0, stream>>>(
            xts[l], wbt, pb[l], feats, nullptr, nullptr,
            M, CC, cins[l], hwsl[l], startl[l], TOK, CC);
    }

    // ---- bottom-up 3x3 stride-2 convs (levels 3 & 4) ----------------------
    { int tot = BB * CC * 8 * 8;
      k_conv3x3s2<<<(tot + 255) / 256, 256, 0, stream>>>(in2, buw0, bub0, m0, 2048, 16, 8);
      k_scatter<<<(tot + 255) / 256, 256, 0, stream>>>(m0, feats, 64, 5376); }
    k_gn_relu<<<BB * 8, 64, 0, stream>>>(m0, gng, gnb, g0, 64);
    { int tot = BB * CC * 4 * 4;
      k_conv3x3s2<<<(tot + 255) / 256, 256, 0, stream>>>(g0, buw1, bub1, m1, 256, 8, 4);
      k_scatter<<<(tot + 255) / 256, 256, 0, stream>>>(m1, feats, 16, 5440); }

    // ---- 6 deformable encoder layers --------------------------------------
    const unsigned MT = tiles(MTOT);   // 171
    for (int i = 0; i < NLAY; i++) {
        // LN1 -> q (bf16)
        k_layernorm<<<MTOT, 32, 0, stream>>>(feats, ln1g + i * CC, ln1b + i * CC, qbf);

        // offsets: q @ woff + boff  -> [MTOT,320] f32
        k_convT<<<(256 * 320 + 255) / 256, 256, 0, stream>>>(
            woff + (size_t)i * 256 * 320, wbt, 256, 320);
        k_gemm64<false, false, false><<<dim3(tiles(320), MT), gb, 0, stream>>>(
            qbf, wbt, boff + (size_t)i * 320, offb, nullptr, nullptr,
            MTOT, 320, 256, MTOT, 0, 0, 320);

        // attention logits: q @ watt + batt -> softmax(20)
        k_convT<<<(256 * 160 + 255) / 256, 256, 0, stream>>>(
            watt + (size_t)i * 256 * 160, wbt, 256, 160);
        k_gemm64<false, false, false><<<dim3(tiles(160), MT), gb, 0, stream>>>(
            qbf, wbt, batt + (size_t)i * 160, attb, nullptr, nullptr,
            MTOT, 160, 256, MTOT, 0, 0, 160);
        k_softmax20<<<(MTOT * NH_ + 255) / 256, 256, 0, stream>>>(attb);

        // values: q @ wval + bval -> [MTOT,256] f32
        k_convT<<<(256 * 256 + 255) / 256, 256, 0, stream>>>(
            wval + (size_t)i * 256 * 256, wbt, 256, 256);
        k_gemm64<false, false, false><<<dim3(tiles(CC), MT), gb, 0, stream>>>(
            qbf, wbt, bval + (size_t)i * 256, valb, nullptr, nullptr,
            MTOT, CC, 256, MTOT, 0, 0, CC);

        // bilinear deformable sampling -> sampb (bf16)
        k_sample<<<MTOT * NH_, 32, 0, stream>>>(offb, attb, valb, sampb);

        // output proj + residual: feats += sampb @ wout + bout
        k_convT<<<(256 * 256 + 255) / 256, 256, 0, stream>>>(
            wout + (size_t)i * 256 * 256, wbt, 256, 256);
        k_gemm64<false, false, true><<<dim3(tiles(CC), MT), gb, 0, stream>>>(
            sampb, wbt, bout + (size_t)i * 256, feats, nullptr, feats,
            MTOT, CC, 256, MTOT, 0, 0, CC);

        // FFN: LN2 -> relu(h@w1+b1) (bf16) -> feats += h@w2 + b2
        k_layernorm<<<MTOT, 32, 0, stream>>>(feats, ln2g + i * CC, ln2b + i * CC, qbf);
        k_convT<<<(256 * 1024 + 255) / 256, 256, 0, stream>>>(
            w1 + (size_t)i * 256 * 1024, wbt, 256, 1024);
        k_gemm64<true, true, false><<<dim3(tiles(FFN_), MT), gb, 0, stream>>>(
            qbf, wbt, b1 + (size_t)i * 1024, nullptr, hidb, nullptr,
            MTOT, FFN_, 256, MTOT, 0, 0, FFN_);
        k_convT<<<(1024 * 256 + 255) / 256, 256, 0, stream>>>(
            w2 + (size_t)i * 1024 * 256, wbt, 1024, 256);
        k_gemm64<false, false, true><<<dim3(tiles(CC), MT), gb, 0, stream>>>(
            hidb, wbt, b2 + (size_t)i * 256, feats, nullptr, feats,
            MTOT, CC, 1024, MTOT, 0, 0, CC);
    }

    // ---- split back into 5 pyramid maps ----------------------------------
    k_out<<<2048, 256, 0, stream>>>(feats, (float*)d_out);
}
